// NodeEdgeLayer_81123342287183
// MI455X (gfx1250) — matrix operand order repaired and checked by reading, not compiled, not run on hardware
//
#include <hip/hip_runtime.h>
#include <hip/hip_bf16.h>

// N=50000, E=500000, D=128, H=4, HD=32 (both N and E divisible by 16)
#define DD   128
#define HH   4

typedef __attribute__((ext_vector_type(16))) __bf16 v16bf;
typedef __attribute__((ext_vector_type(8)))  float  v8f;

__device__ __forceinline__ v8f wmma_bf16(v16bf a, v16bf b, v8f c) {
  // D = A(16x32 bf16) * B(32x16 bf16) + C(16x16 f32)
  return __builtin_amdgcn_wmma_f32_16x16x32_bf16(false, a, false, b, (short)0, c,
                                                 false, false);
}

__device__ __forceinline__ void bf_split(float f, __bf16& hi, __bf16& lo) {
  hi = (__bf16)f;
  lo = (__bf16)(f - (float)hi);
}

// A-fragment (16x32 bf16) K index for element j, lane-half h (ISA 7.12.2)
__device__ __forceinline__ int kmapA(int j, int h) {
  return (j < 8) ? (h * 8 + j) : (16 + h * 8 + (j - 8));
}

// Force a wave-uniform value into an SGPR so guards become SALU branches,
// not exec-mask dances.
__device__ __forceinline__ int wave_uniform(int v) {
  return __builtin_amdgcn_readfirstlane(v);
}

// ---------------------------------------------------------------------------
// Pack fp32 weight matrix W[K x 128] into WMMA B-fragment layout, split into
// bf16 hi/lo pairs.  B (32x16 bf16): lane-half holds K = kmapA(j, half).
// One element per thread: t = ((kt*8 + nt)*32 + lane)*16 + j
// ---------------------------------------------------------------------------
__global__ void k_prep_frags(const float* __restrict__ W, int kt_count,
                             __bf16* __restrict__ fhi, __bf16* __restrict__ flo) {
  int t = blockIdx.x * blockDim.x + threadIdx.x;
  int total = kt_count * 4096;               // kt*8nt*32lane*16j
  if (t >= total) return;
  int j    = t & 15;
  int lane = (t >> 4) & 31;
  int nt   = (t >> 9) & 7;
  int kt   = t >> 12;
  int half = lane >> 4;
  int n    = nt * 16 + (lane & 15);
  int K    = kt * 32 + kmapA(j, half);
  __bf16 hi, lo;
  bf_split(W[(long)K * DD + n], hi, lo);
  fhi[t] = hi;
  flo[t] = lo;
}

// ---------------------------------------------------------------------------
// Node linear: xl = x@W_l + b_l ; xr = x@W_r + b_r.  One wave per 16-row tile.
// ---------------------------------------------------------------------------
__global__ __launch_bounds__(128) void k_node_gemm(
    const float* __restrict__ x, const float* __restrict__ bl,
    const float* __restrict__ br,
    const __bf16* __restrict__ wlh, const __bf16* __restrict__ wll,
    const __bf16* __restrict__ wrh, const __bf16* __restrict__ wrl,
    float* __restrict__ xl, float* __restrict__ xr, int nTiles, int Nn) {
  int lane = threadIdx.x & 31;
  int tile = wave_uniform(blockIdx.x * 4 + (threadIdx.x >> 5));
  if (tile >= nTiles) return;
  const bool full = wave_uniform((tile + 1) * 16 <= Nn ? 1 : 0);
  int half = lane >> 4, r = lane & 15;
  int row  = tile * 16 + r;
  int lrow = (full || row < Nn) ? row : Nn - 1;
  const float* xrow = x + (long)lrow * DD;

  v16bf ahi[4], alo[4];
#pragma unroll
  for (int kc = 0; kc < 4; ++kc)
#pragma unroll
    for (int j = 0; j < 16; ++j) {
      __bf16 hi, lo;
      bf_split(xrow[kc * 32 + kmapA(j, half)], hi, lo);
      ahi[kc][j] = hi;
      alo[kc][j] = lo;
    }

  for (int m = 0; m < 2; ++m) {
    const __bf16* fh  = m ? wrh : wlh;
    const __bf16* fl  = m ? wrl : wll;
    const float*  bia = m ? br : bl;
    float*        out = m ? xr : xl;
#pragma unroll
    for (int nt = 0; nt < 8; ++nt) {
      int   n  = nt * 16 + r;
      float bv = bia[n];
      // preload all B fragments for this column tile (issues as one clause)
      v16bf bh[4], blf[4];
#pragma unroll
      for (int kc = 0; kc < 4; ++kc) {
        bh[kc]  = *(const v16bf*)(fh + (((kc * 8 + nt) * 32 + lane) << 4));
        blf[kc] = *(const v16bf*)(fl + (((kc * 8 + nt) * 32 + lane) << 4));
      }
      // two independent accumulation chains
      v8f acc0, acc1;
#pragma unroll
      for (int i = 0; i < 8; ++i) { acc0[i] = bv; acc1[i] = 0.0f; }
#pragma unroll
      for (int kc = 0; kc < 4; ++kc) {
        acc0 = wmma_bf16(ahi[kc], bh[kc], acc0);
        acc1 = wmma_bf16(alo[kc], bh[kc], acc1);
        acc1 = wmma_bf16(ahi[kc], blf[kc], acc1);
      }
      float* op = out + (long)(tile * 16 + half * 8) * DD + n;
      if (full) {   // scalar branch: no exec masking in the hot path
#pragma unroll
        for (int i = 0; i < 8; ++i) op[(long)i * DD] = acc0[i] + acc1[i];
      } else {
#pragma unroll
        for (int i = 0; i < 8; ++i) {
          int orow = tile * 16 + half * 8 + i;
          if (orow < Nn) out[(long)orow * DD + n] = acc0[i] + acc1[i];
        }
      }
    }
  }
}

// ---------------------------------------------------------------------------
// Init: x_new = bias (agg accumulates via atomics later); emax = enc(-inf);
// denom = 0.
// ---------------------------------------------------------------------------
__global__ void k_init(float* __restrict__ xnew, const float* __restrict__ bias,
                       unsigned* __restrict__ emax, float* __restrict__ denom,
                       long nD, long nH) {
  long t = blockIdx.x * (long)blockDim.x + threadIdx.x;
  if (t < nD) xnew[t] = bias[t & (DD - 1)];
  if (t < nH) {
    emax[t]  = 0x007FFFFFu;  // order-encoded -inf
    denom[t] = 0.0f;
  }
}

// ---------------------------------------------------------------------------
// Edge logits, fused: ee = edge_attr@W_e + b_e (WMMA, never materialized),
// m = ee + xl[src] + xr[dst]; e[E,H] = sum_hd(leaky_relu(m)*att).
// One wave per 16-edge tile; processed per head (2 nt tiles) to limit live registers.
// ---------------------------------------------------------------------------
__global__ __launch_bounds__(128) void k_edge_logits(
    const float* __restrict__ ea, const int* __restrict__ src,
    const int* __restrict__ dst, const __bf16* __restrict__ weh,
    const __bf16* __restrict__ wel, const float* __restrict__ be,
    const float* __restrict__ att, const float* __restrict__ xl,
    const float* __restrict__ xr, float* __restrict__ ebuf, int nTiles,
    int Ee) {
  int lane = threadIdx.x & 31;
  int tile = wave_uniform(blockIdx.x * 4 + (threadIdx.x >> 5));
  if (tile >= nTiles) return;
  const bool full = wave_uniform((tile + 1) * 16 <= Ee ? 1 : 0);
  int half = lane >> 4, r = lane & 15;
  int row  = tile * 16 + r;
  int lrow = (full || row < Ee) ? row : Ee - 1;
  const float* arow = ea + (long)lrow * DD;

  v16bf ahi[4], alo[4];
#pragma unroll
  for (int kc = 0; kc < 4; ++kc)
#pragma unroll
    for (int j = 0; j < 16; ++j) {
      __bf16 hi, lo;
      bf_split(arow[kc * 32 + kmapA(j, half)], hi, lo);
      ahi[kc][j] = hi;
      alo[kc][j] = lo;
    }

  int ebase = tile * 16 + half * 8;
  // hoisted gather indices (uniform across the 16 lanes of each half)
  int sidx[8], didx[8];
#pragma unroll
  for (int i = 0; i < 8; ++i) {
    int ei = ebase + i;
    int ec = (full || ei < Ee) ? ei : Ee - 1;
    sidx[i] = src[ec];
    didx[i] = dst[ec];
  }

  for (int h = 0; h < 4; ++h) {
    float contrib[2][8];
#pragma unroll
    for (int p = 0; p < 2; ++p) {
      int   nt = 2 * h + p;
      int   n  = nt * 16 + r;
      float bv = be[n];
      v16bf bh[4], blf[4];
#pragma unroll
      for (int kc = 0; kc < 4; ++kc) {
        bh[kc]  = *(const v16bf*)(weh + (((kc * 8 + nt) * 32 + lane) << 4));
        blf[kc] = *(const v16bf*)(wel + (((kc * 8 + nt) * 32 + lane) << 4));
      }
      v8f acc0, acc1;
#pragma unroll
      for (int i = 0; i < 8; ++i) { acc0[i] = bv; acc1[i] = 0.0f; }
#pragma unroll
      for (int kc = 0; kc < 4; ++kc) {
        acc0 = wmma_bf16(ahi[kc], bh[kc], acc0);
        acc1 = wmma_bf16(alo[kc], bh[kc], acc1);
        acc1 = wmma_bf16(ahi[kc], blf[kc], acc1);
      }
      float av = att[n];
#pragma unroll
      for (int i = 0; i < 8; ++i) {
        float v = acc0[i] + acc1[i] + xl[(long)sidx[i] * DD + n] +
                  xr[(long)didx[i] * DD + n];
        v = (v >= 0.0f) ? v : 0.2f * v;  // leaky_relu(0.2)
        contrib[p][i] = v * av;
      }
    }
    // reduce 32 columns (2 nt tiles x 16 lanes) per (row, head)
#pragma unroll
    for (int i = 0; i < 8; ++i) {
      float t = contrib[0][i] + contrib[1][i];
      t += __shfl_xor(t, 1, 32);
      t += __shfl_xor(t, 2, 32);
      t += __shfl_xor(t, 4, 32);
      t += __shfl_xor(t, 8, 32);
      int ei = ebase + i;
      if (r == 0 && (full || ei < Ee)) ebuf[(long)ei * HH + h] = t;
    }
  }
}

// ---------------------------------------------------------------------------
// Segment max via order-preserving uint atomicMax.
// ---------------------------------------------------------------------------
__global__ void k_emax(const float* __restrict__ ebuf,
                       const int* __restrict__ dst,
                       unsigned* __restrict__ emax, long total) {
  long t = blockIdx.x * (long)blockDim.x + threadIdx.x;
  if (t >= total) return;
  int  h = (int)(t & 3);
  long e = t >> 2;
  unsigned u    = __float_as_uint(ebuf[t]);
  unsigned mask = (u >> 31) ? 0xFFFFFFFFu : 0x80000000u;
  atomicMax(&emax[(long)dst[e] * HH + h], u ^ mask);
}

__device__ __forceinline__ float dec_key(unsigned k) {
  unsigned mask = (k >> 31) ? 0x80000000u : 0xFFFFFFFFu;
  return __uint_as_float(k ^ mask);
}

// ---------------------------------------------------------------------------
// ex = exp(e - emax[dst]); denom[dst] += ex (segment sum); ex stored in place.
// ---------------------------------------------------------------------------
__global__ void k_expsum(float* __restrict__ ebuf, const int* __restrict__ dst,
                         const unsigned* __restrict__ emax,
                         float* __restrict__ denom, long total) {
  long t = blockIdx.x * (long)blockDim.x + threadIdx.x;
  if (t >= total) return;
  int  h = (int)(t & 3);
  long e = t >> 2;
  int  d = dst[e];
  float m = dec_key(emax[(long)d * HH + h]);
  if (!isfinite(m)) m = 0.0f;  // matches where(isfinite(emax), emax, 0)
  float ex = expf(ebuf[t] - m);
  ebuf[t]  = ex;
  atomicAdd(&denom[(long)d * HH + h], ex);
}

// ---------------------------------------------------------------------------
// x_new[dst] += alpha * xl[src].  One wave per edge, 4 columns per lane.
// ---------------------------------------------------------------------------
__global__ __launch_bounds__(256) void k_agg(
    const float* __restrict__ xl, const int* __restrict__ src,
    const int* __restrict__ dst, const float* __restrict__ ebuf,
    const float* __restrict__ denom, float* __restrict__ xnew, int Ee) {
  int lane = threadIdx.x & 31;
  int edge = blockIdx.x * 8 + (threadIdx.x >> 5);
  if (edge >= Ee) return;
  int s = src[edge], d = dst[edge];
  int c0 = lane * 4;
  int h  = lane >> 3;  // (lane*4)>>5
  float alpha = ebuf[(long)edge * HH + h] / (denom[(long)d * HH + h] + 1e-16f);
  float4 xs = *(const float4*)(xl + (long)s * DD + c0);
  float* o  = xnew + (long)d * DD + c0;
  atomicAdd(o + 0, alpha * xs.x);
  atomicAdd(o + 1, alpha * xs.y);
  atomicAdd(o + 2, alpha * xs.z);
  atomicAdd(o + 3, alpha * xs.w);
}

// ---------------------------------------------------------------------------
// Edge MLP: cat = [x_new[src], x_new[dst]] -> LayerNorm -> ReLU -> @W_mlp
// + b_mlp + edge_attr (residual).  One wave per 16-edge tile; cat tile staged
// in LDS (16x256 f32 = 16KB/wave, 2 waves/block = 32KB).
// ---------------------------------------------------------------------------
__global__ __launch_bounds__(64) void k_edge_mlp(
    const float* __restrict__ xnew, const int* __restrict__ src,
    const int* __restrict__ dst, const float* __restrict__ ea,
    const float* __restrict__ lng, const float* __restrict__ lnb,
    const __bf16* __restrict__ wmh, const __bf16* __restrict__ wml,
    const float* __restrict__ bm, float* __restrict__ eout, int nTiles,
    int Ee) {
  __shared__ __align__(16) float cat[2][16][256];
  int w = threadIdx.x >> 5, lane = threadIdx.x & 31;
  int tile = wave_uniform(blockIdx.x * 2 + (threadIdx.x >> 5));
  const bool active = wave_uniform(tile < nTiles ? 1 : 0);
  const bool full =
      wave_uniform((active && (tile + 1) * 16 <= Ee) ? 1 : 0);

  if (active) {  // gather phase: 1024 float4 slots, 32 lanes
    for (int slot = lane; slot < 16 * 64; slot += 32) {
      int rrow = slot >> 6, q = slot & 63;
      int col  = q * 4;
      int ei   = tile * 16 + rrow;
      int ec   = (full || ei < Ee) ? ei : Ee - 1;
      int node = (col < 128) ? src[ec] : dst[ec];
      float4 v = *(const float4*)(xnew + (long)node * DD + (col & 127));
      *(float4*)&cat[w][rrow][col] = v;
    }
  }
  __syncthreads();

  int half = lane >> 4, r = lane & 15;
  if (active) {  // LayerNorm + ReLU over 256, two lanes per row
    float s = 0.0f, sq = 0.0f;
    float* rowp = &cat[w][r][half * 128];
    for (int c = 0; c < 128; c += 4) {
      float4 v = *(float4*)(rowp + c);
      s  += v.x + v.y + v.z + v.w;
      sq += v.x * v.x + v.y * v.y + v.z * v.z + v.w * v.w;
    }
    s  += __shfl_xor(s, 16, 32);
    sq += __shfl_xor(sq, 16, 32);
    float mean = s * (1.0f / 256.0f);
    float var  = sq * (1.0f / 256.0f) - mean * mean;
    float rstd = rsqrtf(var + 1e-5f);
    const float* g = lng + half * 128;
    const float* b = lnb + half * 128;
    for (int c = 0; c < 128; ++c) {
      float v = (rowp[c] - mean) * rstd * g[c] + b[c];
      rowp[c] = v > 0.0f ? v : 0.0f;
    }
  }
  __syncthreads();
  if (!active) return;

  // GEMM: 16x256 (LDS) @ 256x128 (packed W_mlp), residual in acc init
  v16bf ahi[8], alo[8];
#pragma unroll
  for (int kc = 0; kc < 8; ++kc)
#pragma unroll
    for (int j = 0; j < 16; ++j) {
      __bf16 hi, lo;
      bf_split(cat[w][r][kc * 32 + kmapA(j, half)], hi, lo);
      ahi[kc][j] = hi;
      alo[kc][j] = lo;
    }

  int ebase = tile * 16 + half * 8;
  for (int nt = 0; nt < 8; ++nt) {
    int   n  = nt * 16 + r;
    float bv = bm[n];
    v8f acc0, acc1;
    const float* rp = ea + (long)ebase * DD + n;
#pragma unroll
    for (int i = 0; i < 8; ++i) {
      int ei = ebase + i;
      acc0[i] = bv + ((full || ei < Ee) ? rp[(long)i * DD]
                                        : ea[(long)(Ee - 1) * DD + n]);
      acc1[i] = 0.0f;
    }
    // preload B fragments in two clause groups of 4 kc each
#pragma unroll
    for (int kg = 0; kg < 2; ++kg) {
      v16bf bh[4], blf[4];
#pragma unroll
      for (int k = 0; k < 4; ++k) {
        int kc = kg * 4 + k;
        bh[k]  = *(const v16bf*)(wmh + (((kc * 8 + nt) * 32 + lane) << 4));
        blf[k] = *(const v16bf*)(wml + (((kc * 8 + nt) * 32 + lane) << 4));
      }
#pragma unroll
      for (int k = 0; k < 4; ++k) {
        int kc = kg * 4 + k;
        acc0 = wmma_bf16(ahi[kc], bh[k], acc0);
        acc1 = wmma_bf16(alo[kc], bh[k], acc1);
        acc1 = wmma_bf16(ahi[kc], blf[k], acc1);
      }
    }
    float* op = eout + (long)ebase * DD + n;
    if (full) {   // scalar branch
#pragma unroll
      for (int i = 0; i < 8; ++i) op[(long)i * DD] = acc0[i] + acc1[i];
    } else {
#pragma unroll
      for (int i = 0; i < 8; ++i) {
        int ei = ebase + i;
        if (ei < Ee) eout[(long)ei * DD + n] = acc0[i] + acc1[i];
      }
    }
  }
}

// ---------------------------------------------------------------------------
extern "C" void kernel_launch(void* const* d_in, const int* in_sizes, int n_in,
                              void* d_out, int out_size, void* d_ws,
                              size_t ws_size, hipStream_t stream) {
  (void)n_in; (void)out_size; (void)ws_size;
  const float* x    = (const float*)d_in[0];
  const int*   eidx = (const int*)d_in[1];
  const float* ea   = (const float*)d_in[2];
  const float* Wl   = (const float*)d_in[3];
  const float* bl   = (const float*)d_in[4];
  const float* Wr   = (const float*)d_in[5];
  const float* br   = (const float*)d_in[6];
  const float* We   = (const float*)d_in[7];
  const float* be   = (const float*)d_in[8];
  const float* att  = (const float*)d_in[9];
  const float* bias = (const float*)d_in[10];
  const float* lng  = (const float*)d_in[11];
  const float* lnb  = (const float*)d_in[12];
  const float* Wm   = (const float*)d_in[13];
  const float* bm   = (const float*)d_in[14];

  const long Nn = in_sizes[0] / DD;        // 50000
  const long Ee = in_sizes[2] / DD;        // 500000
  const int* src = eidx;
  const int* dst = eidx + Ee;

  float* xnew = (float*)d_out;             // [N,128]
  float* eout = xnew + Nn * DD;            // [E,128]

  // workspace carve-out (~61 MB)
  char* ws = (char*)d_ws;
  float*    xl    = (float*)ws;    ws += Nn * DD * sizeof(float);
  float*    xr    = (float*)ws;    ws += Nn * DD * sizeof(float);
  float*    ebuf  = (float*)ws;    ws += Ee * HH * sizeof(float);
  unsigned* emax  = (unsigned*)ws; ws += Nn * HH * sizeof(unsigned);
  float*    denom = (float*)ws;    ws += Nn * HH * sizeof(float);
  __bf16* wlh = (__bf16*)ws; ws += 16384 * sizeof(__bf16);
  __bf16* wll = (__bf16*)ws; ws += 16384 * sizeof(__bf16);
  __bf16* wrh = (__bf16*)ws; ws += 16384 * sizeof(__bf16);
  __bf16* wrl = (__bf16*)ws; ws += 16384 * sizeof(__bf16);
  __bf16* weh = (__bf16*)ws; ws += 16384 * sizeof(__bf16);
  __bf16* wel = (__bf16*)ws; ws += 16384 * sizeof(__bf16);
  __bf16* wmh = (__bf16*)ws; ws += 32768 * sizeof(__bf16);
  __bf16* wml = (__bf16*)ws; ws += 32768 * sizeof(__bf16);

  // 1) pack weights into WMMA B-fragment bf16 hi/lo
  k_prep_frags<<<64, 256, 0, stream>>>(Wl, 4, wlh, wll);
  k_prep_frags<<<64, 256, 0, stream>>>(Wr, 4, wrh, wrl);
  k_prep_frags<<<64, 256, 0, stream>>>(We, 4, weh, wel);
  k_prep_frags<<<128, 256, 0, stream>>>(Wm, 8, wmh, wml);

  // 2) node linears (xl, xr)
  int nTilesN = (int)((Nn + 15) / 16);
  k_node_gemm<<<(nTilesN + 3) / 4, 128, 0, stream>>>(
      x, bl, br, wlh, wll, wrh, wrl, xl, xr, nTilesN, (int)Nn);

  // 3) init accumulators / x_new = bias
  long nD = Nn * DD, nH = Nn * HH;
  k_init<<<(int)((nD + 255) / 256), 256, 0, stream>>>(xnew, bias, emax, denom,
                                                      nD, nH);

  // 4) fused edge GEMM -> attention logits
  int nTilesE = (int)((Ee + 15) / 16);
  k_edge_logits<<<(nTilesE + 3) / 4, 128, 0, stream>>>(
      ea, src, dst, weh, wel, be, att, xl, xr, ebuf, nTilesE, (int)Ee);

  // 5) segment softmax (max, exp, sum)
  long tot = Ee * HH;
  k_emax<<<(int)((tot + 255) / 256), 256, 0, stream>>>(ebuf, dst, emax, tot);
  k_expsum<<<(int)((tot + 255) / 256), 256, 0, stream>>>(ebuf, dst, emax, denom,
                                                         tot);

  // 6) weighted aggregation into x_new
  k_agg<<<(int)((Ee + 7) / 8), 256, 0, stream>>>(xl, src, dst, ebuf, denom,
                                                 xnew, (int)Ee);

  // 7) edge MLP with residual
  k_edge_mlp<<<(nTilesE + 1) / 2, 64, 0, stream>>>(
      xnew, src, dst, ea, lng, lnb, wmh, wml, bm, eout, nTilesE, (int)Ee);
}
